// STC_Cell_16647293239398
// MI455X (gfx1250) — matrix operand order repaired and checked by reading, not compiled, not run on hardware
//
#include <hip/hip_runtime.h>
#include <hip/hip_bf16.h>
#include <math.h>

// ---------------------------------------------------------------------------
// STC-Cell (graph-conv GRU) for MI455X / gfx1250, wave32.
// All heavy math on v_wmma_f32_16x16x32_bf16 with bf16 hi/lo split (bf16x3)
// for ~fp32 accuracy at bf16 matrix-pipe rates.
//   Pass A: T2 = 2*Gs*Gs - I             (2048x2048x2048 GEMM)
//   Pass B: spatial conv X1_k = Gs_k^T X (2 GEMMs 2048x2048x8448 per bdg_dif)
//   Pass C: fused channel-mix+projection: one GEMM 32768x1600x{1024,512}
//           with folded weights W2[(k,c,l),(d,h)] = sum_j Gc_st[j,c,d]*W[(k,j,l),h]
//   GRU nonlinearities fused into projection epilogues.
// v2: all B operands stored pre-transposed ([col][K]) so LDS staging is pure
//     b128 copies (no scalar transpose), double-buffered LDS (one barrier per
//     K-step, global loads of tile k+1 overlap WMMAs of tile k), plus
//     global_prefetch of tile k+2.
// Workspace budget ~686 MB.
// ---------------------------------------------------------------------------

typedef __attribute__((ext_vector_type(16))) __bf16 v16bf;
typedef __attribute__((ext_vector_type(8)))  float  v8f;

#define B_     16
#define N_     2048
#define C_     8
#define L_     66       // Din + H
#define H_     64
#define KCL_   1584     // KS*C*L
#define KP_    1600     // padded to multiple of 32
#define JCOLS_ 8448     // B*C*L

constexpr int BM = 128, BN = 128, BK = 32, BKP = BK + 8;

union FragU { v16bf v; __bf16 e[16]; };
union AccU  { v8f   v; float  e[8];  };

__device__ inline void bsplit(float x, __bf16* hi, __bf16* lo) {
  __bf16 h = (__bf16)x;
  *hi = h;
  *lo = (__bf16)(x - (float)h);
}

struct GemmArgs {
  const __bf16 *Ahi, *Alo;     // A planes, row-major [M][K]
  const __bf16 *Bhi, *Blo;     // B planes, pre-transposed [Ncol][K]
  int lda, ldb, K;
  __bf16 *o0h, *o0l, *o1h, *o1l;   // bf16-split outputs
  float *f0;                        // update buffer (mode2 write / mode3 read)
  float *fout;                      // mode3: final Ht
  const float *bias;
  const float *ht1;
};

// Stage one (A,B) K-tile into LDS: pure 16-byte vector copies for both
// operands (B is already [col][K] in global memory).
__device__ inline void stage_tile(const GemmArgs& P,
                                  const __bf16* Ah, const __bf16* Al,
                                  __bf16* Ash, __bf16* Asl,
                                  __bf16* Bsh, __bf16* Bsl,
                                  int bm0, int bn0, int k0, int tid) {
  const int row = tid >> 1, kk = (tid & 1) * 16;
  {
    const uint4* sh = (const uint4*)(Ah + (size_t)(bm0 + row) * P.lda + k0 + kk);
    const uint4* sl = (const uint4*)(Al + (size_t)(bm0 + row) * P.lda + k0 + kk);
    uint4* dh = (uint4*)&Ash[row * BKP + kk];
    uint4* dl = (uint4*)&Asl[row * BKP + kk];
    dh[0] = sh[0]; dh[1] = sh[1];
    dl[0] = sl[0]; dl[1] = sl[1];
  }
  {
    const uint4* sh = (const uint4*)(P.Bhi + (size_t)(bn0 + row) * P.ldb + k0 + kk);
    const uint4* sl = (const uint4*)(P.Blo + (size_t)(bn0 + row) * P.ldb + k0 + kk);
    uint4* dh = (uint4*)&Bsh[row * BKP + kk];
    uint4* dl = (uint4*)&Bsl[row * BKP + kk];
    dh[0] = sh[0]; dh[1] = sh[1];
    dl[0] = sl[0]; dl[1] = sl[1];
  }
}

// MODE 0: T2=2*Gs*Gs-I, store transposed bf16-split
// MODE 1: spatial conv (grid.z selects Gs^T / T2^T), store into X1 stack
// MODE 2: gates projection + sigmoid; write update buf, candi planes, X1B k0
// MODE 3: candidate projection + tanh + GRU blend -> d_out
template<int MODE>
__global__ __launch_bounds__(256)
void gemm_bf16x3(GemmArgs P) {
  __shared__ __bf16 Ash[2][BM*BKP], Asl[2][BM*BKP], Bsh[2][BN*BKP], Bsl[2][BN*BKP];

  const int tid  = threadIdx.x;
  const int lane = tid & 31;
  const int wid  = tid >> 5;                 // 8 waves: 2 (M) x 4 (N)
  const int m0w  = (wid & 1) * 64;
  const int n0w  = (wid >> 1) * 32;
  const int bm0  = blockIdx.x * BM;
  const int bn0  = blockIdx.y * BN;

  size_t aoff = 0;
  if constexpr (MODE == 1) aoff = (size_t)blockIdx.z * (size_t)N_ * (size_t)N_;
  const __bf16* Ah = P.Ahi + aoff;
  const __bf16* Al = P.Alo + aoff;

  AccU acc[4][2];
#pragma unroll
  for (int i = 0; i < 4; ++i)
#pragma unroll
    for (int j = 0; j < 2; ++j)
#pragma unroll
      for (int e = 0; e < 8; ++e) acc[i][j].e[e] = 0.f;

  const int nk = P.K / BK;
  stage_tile(P, Ah, Al, Ash[0], Asl[0], Bsh[0], Bsl[0], bm0, bn0, 0, tid);

  for (int kt = 0; kt < nk; ++kt) {
    __syncthreads();
    const int cur = kt & 1, nxt = cur ^ 1;
    if (kt + 1 < nk) {
      stage_tile(P, Ah, Al, Ash[nxt], Asl[nxt], Bsh[nxt], Bsl[nxt],
                 bm0, bn0, (kt + 1) * BK, tid);
      if (kt + 2 < nk) {  // prefetch tile k+2 toward L2/L0
        const int row = tid >> 1, kk2 = (tid & 1) * 16 + (kt + 2) * BK;
        __builtin_prefetch(Ah + (size_t)(bm0 + row) * P.lda + kk2, 0, 1);
        __builtin_prefetch(P.Bhi + (size_t)(bn0 + row) * P.ldb + kk2, 0, 1);
        __builtin_prefetch(Al + (size_t)(bm0 + row) * P.lda + kk2, 0, 1);
        __builtin_prefetch(P.Blo + (size_t)(bn0 + row) * P.ldb + kk2, 0, 1);
      }
    }

    // ---- fragments (ISA wave32 16x32 A / 32x16 B layouts) ----
    FragU ah[4], al[4], bh[2], bl[2];
#pragma unroll
    for (int mt = 0; mt < 4; ++mt) {
      int row = m0w + mt * 16 + (lane & 15);
      int kb  = (lane >> 4) * 8;
      const __bf16* ph = &Ash[cur][row * BKP + kb];
      const __bf16* pl = &Asl[cur][row * BKP + kb];
#pragma unroll
      for (int i = 0; i < 8; ++i) {
        ah[mt].e[i] = ph[i];  ah[mt].e[8 + i] = ph[16 + i];
        al[mt].e[i] = pl[i];  al[mt].e[8 + i] = pl[16 + i];
      }
    }
#pragma unroll
    for (int nt = 0; nt < 2; ++nt) {
      int col = n0w + nt * 16 + (lane & 15);
      int kb  = (lane >> 4) * 8;
      const __bf16* ph = &Bsh[cur][col * BKP + kb];
      const __bf16* pl = &Bsl[cur][col * BKP + kb];
#pragma unroll
      for (int i = 0; i < 8; ++i) {
        bh[nt].e[i] = ph[i];  bh[nt].e[8 + i] = ph[16 + i];
        bl[nt].e[i] = pl[i];  bl[nt].e[8 + i] = pl[16 + i];
      }
    }

    // ---- bf16x3: hi*hi + hi*lo + lo*hi ----
#pragma unroll
    for (int mt = 0; mt < 4; ++mt)
#pragma unroll
      for (int nt = 0; nt < 2; ++nt) {
        acc[mt][nt].v = __builtin_amdgcn_wmma_f32_16x16x32_bf16(
            false, ah[mt].v, false, bh[nt].v, (short)0, acc[mt][nt].v, false, false);
        acc[mt][nt].v = __builtin_amdgcn_wmma_f32_16x16x32_bf16(
            false, ah[mt].v, false, bl[nt].v, (short)0, acc[mt][nt].v, false, false);
        acc[mt][nt].v = __builtin_amdgcn_wmma_f32_16x16x32_bf16(
            false, al[mt].v, false, bh[nt].v, (short)0, acc[mt][nt].v, false, false);
      }
  }

  // ---- epilogue (C/D layout: elem r -> row 8*(lane>>4)+r, col lane&15) ----
#pragma unroll
  for (int mt = 0; mt < 4; ++mt)
#pragma unroll
    for (int nt = 0; nt < 2; ++nt)
#pragma unroll
      for (int r = 0; r < 8; ++r) {
        int gm = bm0 + m0w + mt * 16 + ((lane >> 4) << 3) + r;
        int gn = bn0 + n0w + nt * 16 + (lane & 15);
        float v = acc[mt][nt].e[r];
        if constexpr (MODE == 0) {
          v = 2.f * v - ((gm == gn) ? 1.f : 0.f);
          __bf16 hi, lo; bsplit(v, &hi, &lo);
          size_t o = (size_t)gn * N_ + gm;               // store transposed
          P.o0h[o] = hi; P.o0l[o] = lo;
        } else if constexpr (MODE == 1) {
          int b = gn / 528, cl = gn - b * 528;
          int kIdx = (int)blockIdx.z + 1;
          size_t o = ((size_t)b * N_ + gm) * KP_ + kIdx * 528 + cl;
          __bf16 hi, lo; bsplit(v, &hi, &lo);
          P.o0h[o] = hi; P.o0l[o] = lo;
        } else if constexpr (MODE == 2) {
          int g = gn & 127, d = gn >> 7;
          float s = 1.f / (1.f + __expf(-(v + P.bias[g])));
          if (g < 64) {
            P.f0[((size_t)gm * C_ + d) * H_ + g] = s;    // update gate
          } else {
            int h = g - 64;
            float rH = s * P.ht1[((size_t)gm * C_ + d) * H_ + h];
            __bf16 hi, lo; bsplit(rH, &hi, &lo);
            int b = gm >> 11, m = gm & 2047;
            // candi B-planes, [j=(b,c,l)][node] layout
            size_t po = ((size_t)(b * 528 + d * 66 + 2 + h)) * N_ + m;
            P.o0h[po] = hi; P.o0l[po] = lo;
            size_t xo = (size_t)gm * KP_ + d * 66 + 2 + h;  // X1B k=0 slice
            P.o1h[xo] = hi; P.o1l[xo] = lo;
          }
        } else {  // MODE == 3
          int h = gn & 63, d = gn >> 6;
          float c = tanhf(v + P.bias[h]);
          size_t idx = ((size_t)gm * C_ + d) * H_ + h;
          float u  = P.f0[idx];
          float h1 = P.ht1[idx];
          P.fout[idx] = (1.f - u) * h1 + u * c;
        }
      }
}

// ---------- small prep kernels ----------

__global__ void k_gcst(const float* Gc, float* gcst) {
  int t = threadIdx.x;           // 64 threads
  int c = t >> 3, d = t & 7;
  float s = 0.f;
  for (int e = 0; e < 8; ++e) s += Gc[c * 8 + e] * Gc[e * 8 + d];
  gcst[t]        = (c == d) ? 1.f : 0.f;
  gcst[64 + t]   = Gc[t];
  gcst[128 + t]  = 2.f * s - ((c == d) ? 1.f : 0.f);
}

// Folded weights, stored transposed: W2planes[(d*OH+g)][kcl], ld = KP_
__global__ void k_fold_w(const float* gcst, const float* W, int OH,
                         __bf16* oh, __bf16* ol) {
  int idx = blockIdx.x * blockDim.x + threadIdx.x;
  int cols = C_ * OH;
  if (idx >= KP_ * cols) return;
  int kcl = idx % KP_, col = idx / KP_;
  int d = col / OH, g = col - d * OH;
  float v = 0.f;
  if (kcl < KCL_) {
    int k = kcl / 528, r0 = kcl - k * 528;
    int c = r0 / 66, l = r0 - c * 66;
    for (int j = 0; j < 3; ++j)
      v += gcst[j * 64 + c * 8 + d] * W[((k * 3 + j) * 66 + l) * OH + g];
  }
  __bf16 hi, lo; bsplit(v, &hi, &lo);
  oh[idx] = hi; ol[idx] = lo;
}

__global__ void k_prep_gs(const float* Gs, __bf16* gh, __bf16* gl,
                          __bf16* th, __bf16* tl) {
  int idx = blockIdx.x * blockDim.x + threadIdx.x;
  if (idx >= N_ * N_) return;
  int m = idx / N_, n = idx - m * N_;
  __bf16 hi, lo;
  bsplit(Gs[idx], &hi, &lo);                 gh[idx] = hi; gl[idx] = lo;
  bsplit(Gs[(size_t)n * N_ + m], &hi, &lo);  th[idx] = hi; tl[idx] = lo; // Gs^T
}

// XH planes stored [j=(b,c,l)][node], ld = N_  (=> write offset == idx)
__global__ void k_xh(const float* Xt, const float* Ht1,
                     __bf16* xhh, __bf16* xhl, __bf16* cbh, __bf16* cbl,
                     __bf16* x1ah, __bf16* x1al, __bf16* x1bh, __bf16* x1bl) {
  size_t idx = (size_t)blockIdx.x * blockDim.x + threadIdx.x;
  if (idx >= (size_t)JCOLS_ * N_) return;
  int n = (int)(idx & (N_ - 1));
  int j = (int)(idx >> 11);
  int b = j / 528, r0 = j - b * 528;
  int c = r0 / 66, l = r0 - c * 66;
  float v = (l < 2) ? Xt[(((size_t)b * N_ + n) * C_ + c) * 2 + l]
                    : Ht1[(((size_t)b * N_ + n) * C_ + c) * H_ + (l - 2)];
  __bf16 hi, lo; bsplit(v, &hi, &lo);
  xhh[idx] = hi; xhl[idx] = lo;
  size_t xo = ((size_t)b * N_ + n) * KP_ + c * 66 + l;
  x1ah[xo] = hi; x1al[xo] = lo;
  if (l < 2) { cbh[idx] = hi; cbl[idx] = lo; x1bh[xo] = hi; x1bl[xo] = lo; }
}

__global__ void k_zeropad(__bf16* ah, __bf16* al, __bf16* bh2, __bf16* bl2) {
  int idx = blockIdx.x * blockDim.x + threadIdx.x;
  if (idx >= 32768 * 16) return;
  size_t row = (size_t)(idx >> 4);
  size_t o = row * KP_ + KCL_ + (idx & 15);
  ah[o] = (__bf16)0.f; al[o] = (__bf16)0.f;
  bh2[o] = (__bf16)0.f; bl2[o] = (__bf16)0.f;
}

// ---------- launch ----------

extern "C" void kernel_launch(void* const* d_in, const int* in_sizes, int n_in,
                              void* d_out, int out_size, void* d_ws, size_t ws_size,
                              hipStream_t stream) {
  (void)in_sizes; (void)n_in; (void)out_size; (void)ws_size;
  const float* Gs  = (const float*)d_in[0];
  const float* Gc  = (const float*)d_in[1];
  const float* Xt  = (const float*)d_in[2];
  const float* Ht1 = (const float*)d_in[3];
  const float* Wg  = (const float*)d_in[4];
  const float* bg  = (const float*)d_in[5];
  const float* Wc  = (const float*)d_in[6];
  const float* bc  = (const float*)d_in[7];

  char* ws = (char*)d_ws;
  size_t off = 0;
  auto alloc = [&](size_t bytes) -> void* {
    void* p = ws + off;
    off = (off + bytes + 255) & ~(size_t)255;
    return p;
  };

  float*  gcst  = (float*) alloc(192 * 4);
  __bf16* W2gh  = (__bf16*)alloc((size_t)KP_ * 1024 * 2);
  __bf16* W2gl  = (__bf16*)alloc((size_t)KP_ * 1024 * 2);
  __bf16* W2ch  = (__bf16*)alloc((size_t)KP_ * 512 * 2);
  __bf16* W2cl  = (__bf16*)alloc((size_t)KP_ * 512 * 2);
  __bf16* Gsh   = (__bf16*)alloc((size_t)N_ * N_ * 2);
  __bf16* Gsl   = (__bf16*)alloc((size_t)N_ * N_ * 2);
  __bf16* ATh   = (__bf16*)alloc((size_t)2 * N_ * N_ * 2);   // [Gs^T | T2^T]
  __bf16* ATl   = (__bf16*)alloc((size_t)2 * N_ * N_ * 2);
  __bf16* XHh   = (__bf16*)alloc((size_t)N_ * JCOLS_ * 2);
  __bf16* XHl   = (__bf16*)alloc((size_t)N_ * JCOLS_ * 2);
  __bf16* CBh   = (__bf16*)alloc((size_t)N_ * JCOLS_ * 2);
  __bf16* CBl   = (__bf16*)alloc((size_t)N_ * JCOLS_ * 2);
  __bf16* X1Ah  = (__bf16*)alloc((size_t)32768 * KP_ * 2);
  __bf16* X1Al  = (__bf16*)alloc((size_t)32768 * KP_ * 2);
  __bf16* X1Bh  = (__bf16*)alloc((size_t)32768 * KP_ * 2);
  __bf16* X1Bl  = (__bf16*)alloc((size_t)32768 * KP_ * 2);
  float*  updat = (float*) alloc((size_t)32768 * 512 * 4);
  // total ~686 MB

  // 1) Gc Chebyshev stack + folded (transposed) weights
  hipLaunchKernelGGL(k_gcst, dim3(1), dim3(64), 0, stream, Gc, gcst);
  hipLaunchKernelGGL(k_fold_w, dim3((KP_ * 1024 + 255) / 256), dim3(256), 0, stream,
                     gcst, Wg, 128, W2gh, W2gl);
  hipLaunchKernelGGL(k_fold_w, dim3((KP_ * 512 + 255) / 256), dim3(256), 0, stream,
                     gcst, Wc, 64, W2ch, W2cl);

  // 2) Gs splits (plain + transposed)
  hipLaunchKernelGGL(k_prep_gs, dim3((N_ * N_ + 255) / 256), dim3(256), 0, stream,
                     Gs, Gsh, Gsl, ATh, ATl);

  // 3) T2 = 2*Gs*Gs - I  (store T2^T split at ATh+N*N).  Note for this GEMM
  //    B must be [n][k] = Gs^T-as-[col][K]; Gs planes in natural [k][n] order
  //    are exactly that when read as [col=n? ] -- we need B[k][n] with
  //    [col][K] layout == Gs^T planes, which k_prep_gs already produced (ATh).
  {
    GemmArgs P{};
    P.Ahi = Gsh; P.Alo = Gsl;          // A[m][k] = Gs
    P.Bhi = ATh; P.Blo = ATl;          // B [col=n][K=k] = Gs[k][n] = Gs^T plane
    P.lda = N_; P.ldb = N_; P.K = N_;
    P.o0h = ATh + (size_t)N_ * N_; P.o0l = ATl + (size_t)N_ * N_;
    hipLaunchKernelGGL(gemm_bf16x3<0>, dim3(16, 16, 1), dim3(256), 0, stream, P);
  }

  // 4) XH concat -> B-planes pass 1, Xt slice of pass-2 planes, k=0 slices
  hipLaunchKernelGGL(k_xh, dim3((unsigned)(((size_t)JCOLS_ * N_ + 255) / 256)),
                     dim3(256), 0, stream,
                     Xt, Ht1, XHh, XHl, CBh, CBl, X1Ah, X1Al, X1Bh, X1Bl);
  hipLaunchKernelGGL(k_zeropad, dim3((32768 * 16 + 255) / 256), dim3(256), 0, stream,
                     X1Ah, X1Al, X1Bh, X1Bl);

  // 5) spatial conv pass 1 (k=1,2 via grid.z)
  {
    GemmArgs P{};
    P.Ahi = ATh; P.Alo = ATl; P.Bhi = XHh; P.Blo = XHl;
    P.lda = N_; P.ldb = N_; P.K = N_;
    P.o0h = X1Ah; P.o0l = X1Al;
    hipLaunchKernelGGL(gemm_bf16x3<1>, dim3(16, 66, 2), dim3(256), 0, stream, P);
  }

  // 6) gates projection + sigmoid; writes update buf, candi planes, X1B k=0
  {
    GemmArgs P{};
    P.Ahi = X1Ah; P.Alo = X1Al; P.Bhi = W2gh; P.Blo = W2gl;
    P.lda = KP_; P.ldb = KP_; P.K = KP_;
    P.o0h = CBh; P.o0l = CBl; P.o1h = X1Bh; P.o1l = X1Bl;
    P.f0 = updat; P.bias = bg; P.ht1 = Ht1;
    hipLaunchKernelGGL(gemm_bf16x3<2>, dim3(256, 8, 1), dim3(256), 0, stream, P);
  }

  // 7) spatial conv pass 2 over candi_in
  {
    GemmArgs P{};
    P.Ahi = ATh; P.Alo = ATl; P.Bhi = CBh; P.Blo = CBl;
    P.lda = N_; P.ldb = N_; P.K = N_;
    P.o0h = X1Bh; P.o0l = X1Bl;
    hipLaunchKernelGGL(gemm_bf16x3<1>, dim3(16, 66, 2), dim3(256), 0, stream, P);
  }

  // 8) candidate projection + tanh + GRU blend -> Ht
  {
    GemmArgs P{};
    P.Ahi = X1Bh; P.Alo = X1Bl; P.Bhi = W2ch; P.Blo = W2cl;
    P.lda = KP_; P.ldb = KP_; P.K = KP_;
    P.f0 = updat; P.bias = bc; P.ht1 = Ht1; P.fout = (float*)d_out;
    hipLaunchKernelGGL(gemm_bf16x3<3>, dim3(256, 4, 1), dim3(256), 0, stream, P);
  }
}
